// CharRNN_53214644797881
// MI455X (gfx1250) — matrix operand order repaired and checked by reading, not compiled, not run on hardware
//
#include <hip/hip_runtime.h>
#include <hip/hip_bf16.h>

// ---------------------------------------------------------------------------
// CharRNN LSTM forward + CE loss for MI455X (gfx1250, wave32, WMMA).
//
// Sizes: B=8, T=2048, V=256, H=1024, 4H=4096. Output: scalar mean NLL.
//
// Roofline: 146 GFLOP total, ~100 MB unique traffic (4 us at 23.3 TB/s) ->
// bound by the T=2048 serial recurrence. Strategy: batch all 8 sequences into
// one N=16 WMMA tile (W_hh bf16 = 8MB, resident in 192MB L2, read once per
// step), 32 persistent WGs each owning 32 hidden rows (128 interleaved gate
// rows) so the cell update is WG-local and ONE global barrier per timestep
// suffices (double-buffered h). h staged to LDS via async global->LDS.
// ---------------------------------------------------------------------------

static constexpr int kV = 256;
static constexpr int kH = 1024;
static constexpr int kB = 8;
static constexpr int kT = 2048;
static constexpr int kG = 4 * kH;          // 4096 gate rows
static constexpr int kNWG = 32;            // persistent workgroups (recurrent)
static constexpr int kNT = kB * kT;        // 16384 (b,t) positions

typedef __attribute__((ext_vector_type(16))) __bf16 v16bf;
typedef __attribute__((ext_vector_type(8)))  __bf16 v8bf;
typedef __attribute__((ext_vector_type(8)))  float  v8f;
typedef __attribute__((ext_vector_type(4)))  int    v4i;

// workspace layout (bytes)
static constexpr size_t CNT_OFF  = 0;                                  // barrier counter
static constexpr size_t HBUF_OFF = 1024;                               // h double buffer [2][16][H] bf16 (64KB)
static constexpr size_t WHH_OFF  = 128 * 1024;                         // W_hh bf16 [4096][1024] (8MB)
static constexpr size_t W1_OFF   = WHH_OFF + (size_t)kG * kH * 2;      // W1 bf16 [256][1024] (512KB)
static constexpr size_t HS_OFF   = W1_OFF + (size_t)kV * kH * 2;       // hs bf16 [16384][1024] (32MB)
static constexpr size_t WS_NEED  = HS_OFF + (size_t)kNT * kH * 2;

// Activations built on v_exp_f32 + v_rcp_f32 (no precise-division expansion
// on the recurrence critical path).
__device__ __forceinline__ float sigmoidf_fast(float x) {
  return __builtin_amdgcn_rcpf(1.0f + __expf(-x));
}
__device__ __forceinline__ float tanhf_fast(float x) {
  float e = __expf(-2.0f * fabsf(x));
  float t = (1.0f - e) * __builtin_amdgcn_rcpf(1.0f + e);
  return copysignf(t, x);
}

// Build a v16bf A-fragment (contiguous K order, matching B):
// lane m=l&15, half=l>>4; elements 0..15: K = half*16 + 0..15
__device__ __forceinline__ v16bf load_a_frag(const __bf16* row, int k0, int half) {
  return *(const v16bf*)(row + k0 + half * 16);
}

// Stage 16*kH bf16 (32KB) global -> LDS. Prefer gfx1250 async global->LDS
// (no VGPR round trip, tracked by ASYNCcnt); fall back to a plain copy.
__device__ __forceinline__ void stage_32k(const __bf16* __restrict__ src,
                                          __bf16* __restrict__ dst, int tid) {
#if __has_builtin(__builtin_amdgcn_global_load_async_to_lds_b128)
  typedef __attribute__((address_space(1))) v4i* gv4i_p;   // global int4*
  typedef __attribute__((address_space(3))) v4i* lv4i_p;   // LDS int4*
  #pragma unroll
  for (int i = 0; i < (16 * kH) / 8 / 256; ++i) {   // 8 x 16B per thread
    const __bf16* s = src + (size_t)(tid + i * 256) * 8;
    __bf16*       d = dst + (size_t)(tid + i * 256) * 8;
    __builtin_amdgcn_global_load_async_to_lds_b128((gv4i_p)s, (lv4i_p)d, 0, 0);
  }
#if __has_builtin(__builtin_amdgcn_s_wait_asynccnt)
  __builtin_amdgcn_s_wait_asynccnt(0);
#else
  asm volatile("s_wait_asynccnt 0x0" ::: "memory");
#endif
#else
  const uint4* s4 = (const uint4*)src;
  uint4*       d4 = (uint4*)dst;
  #pragma unroll
  for (int i = 0; i < (16 * kH) / 8 / 256; ++i)
    d4[tid + i * 256] = s4[tid + i * 256];
#endif
}

// ---------------------------------------------------------------------------
// Init: zero scalar output, barrier counter, h double-buffer; convert weights
// to bf16. Must run every launch (buffers are poisoned, no cross-call state).
// ---------------------------------------------------------------------------
__global__ void k_init(const float* __restrict__ Whh, const float* __restrict__ W1,
                       float* __restrict__ out, unsigned* __restrict__ cnt,
                       __bf16* __restrict__ hbuf, __bf16* __restrict__ Whh_bf,
                       __bf16* __restrict__ W1_bf) {
  size_t tid = (size_t)blockIdx.x * blockDim.x + threadIdx.x;
  size_t stride = (size_t)gridDim.x * blockDim.x;
  if (tid == 0) out[0] = 0.0f;
  if (tid < 8) cnt[tid] = 0u;
  for (size_t i = tid; i < (size_t)2 * 16 * kH; i += stride) hbuf[i] = (__bf16)0.0f;
  for (size_t i = tid; i < (size_t)kG * kH; i += stride) Whh_bf[i] = (__bf16)Whh[i];
  for (size_t i = tid; i < (size_t)kV * kH; i += stride) W1_bf[i] = (__bf16)W1[i];
}

// ---------------------------------------------------------------------------
// Persistent recurrent kernel: 32 WGs x 256 threads (8 wave32s).
// WG wg owns hidden rows [wg*32, wg*32+32). Wave w computes the 16x16 tile of
// gate rows  blk*1024 + wg*32 + sub*16 .. +15  (blk=w>>1 in {i,f,g,o}, sub=w&1)
// against h[1024 x 16] (cols 0..7 = batch, 8..15 = zero pad).
// ---------------------------------------------------------------------------
__global__ void __launch_bounds__(256, 1)
k_lstm(const int* __restrict__ Xs, const float* __restrict__ W_ih,
       const float* __restrict__ b_ih, const float* __restrict__ b_hh,
       const __bf16* __restrict__ Whh_bf, __bf16* __restrict__ hbuf,
       __bf16* __restrict__ hs, unsigned* __restrict__ cnt) {
  __shared__ __align__(16) __bf16 h_lds[16 * kH];         // B matrix, [n][k], 32KB
  __shared__ __align__(16) float  gates_lds[4 * 32 * 16]; // [blk][lr][n], 8KB
  __shared__ float c_lds[32 * 8];                         // cell state, 1KB

  const int tid  = threadIdx.x;
  const int wg   = blockIdx.x;          // 0..31
  const int wave = tid >> 5;            // 0..7
  const int lane = tid & 31;
  const int half = lane >> 4;
  const int m    = lane & 15;           // A-row index
  const int n    = lane & 15;           // B/C column (batch) index

  const int blk = wave >> 1;            // gate block: 0=i 1=f 2=g 3=o
  const int sub = wave & 1;
  const int rowbase = blk * kH + wg * 32 + sub * 16;

  for (int i = tid; i < 32 * 8; i += 256) c_lds[i] = 0.0f;

  const __bf16* aptr = Whh_bf + (size_t)(rowbase + m) * kH;   // this lane's A row

  // t-invariant pieces of the xg add: mask, clamped Xs row, biases, W_ih base.
  const float msk  = (n < kB) ? 1.0f : 0.0f;
  const int   xrow = (n & 7) * kT;                            // always in-bounds
  const float* wih = W_ih + (size_t)(rowbase + half * 8) * kV;
  float bias_r[8];
  #pragma unroll
  for (int r = 0; r < 8; ++r) {
    const int row = rowbase + half * 8 + r;
    bias_r[r] = b_ih[row] + b_hh[row];
  }
  const int gbase = blk * 512 + (sub * 16 + half * 8) * 16 + n;

  for (int t = 0; t < kT; ++t) {
    // Stage h (step t) into LDS via async global->LDS from the read buffer.
    stage_32k(hbuf + (size_t)(t & 1) * 16 * kH, h_lds, tid);

    // Issue the one-hot W_ih gather early so it overlaps the WMMA K-loop.
    const int x = Xs[xrow + t];
    float wih_r[8];
    #pragma unroll
    for (int r = 0; r < 8; ++r) wih_r[r] = wih[(size_t)r * kV + x];

    __syncthreads();

    // gates_tile = Whh_slice[16x1024] @ h[1024x16]   (bf16 WMMA, f32 acc)
    v8f acc = {};
    #pragma unroll 4
    for (int kb = 0; kb < kH / 32; ++kb) {
      const int k0 = kb * 32;
      v16bf a = load_a_frag(aptr, k0, half);
      // B layout: lane n=l&15, half=l>>4 holds K = k0 + half*16 + 0..15 (contiguous)
      v16bf b = *(const v16bf*)(h_lds + (size_t)n * kH + k0 + half * 16);
      acc = __builtin_amdgcn_wmma_f32_16x16x32_bf16(false, a, false, b,
                                                    (short)0, acc, false, false);
    }

    // Branch-free xg + bias add (masked FMA), spill gate tile to LDS.
    #pragma unroll
    for (int r = 0; r < 8; ++r)
      gates_lds[gbase + r * 16] = fmaf(msk, wih_r[r] + bias_r[r], acc[r]);
    __syncthreads();

    // LSTM cell update: 256 threads <-> 32 local rows x 8 batch.
    {
      const int lr = tid >> 3;           // local hidden row
      const int nb = tid & 7;            // batch
      float iv = sigmoidf_fast(gates_lds[0 * 512 + lr * 16 + nb]);
      float fv = sigmoidf_fast(gates_lds[1 * 512 + lr * 16 + nb]);
      float gv = tanhf_fast   (gates_lds[2 * 512 + lr * 16 + nb]);
      float ov = sigmoidf_fast(gates_lds[3 * 512 + lr * 16 + nb]);
      float c  = fv * c_lds[lr * 8 + nb] + iv * gv;
      c_lds[lr * 8 + nb] = c;
      float h = ov * tanhf_fast(c);
      const int j = wg * 32 + lr;
      hbuf[(size_t)((t + 1) & 1) * 16 * kH + (size_t)nb * kH + j] = (__bf16)h;  // write buffer
      hs[((size_t)nb * kT + t) * kH + j] = (__bf16)h;                           // for projection
    }

    // One global barrier per step (monotonic counter; reset each launch).
    __syncthreads();
    if (tid == 0) {
      __threadfence();
      atomicAdd(cnt, 1u);
      const unsigned target = (unsigned)(t + 1) * kNWG;
      while (__hip_atomic_load(cnt, __ATOMIC_RELAXED, __HIP_MEMORY_SCOPE_AGENT) < target)
        __builtin_amdgcn_s_sleep(1);
    }
    __syncthreads();
    __threadfence();
  }
}

// ---------------------------------------------------------------------------
// Output projection + cross-entropy: 1024 WGs, one 16-column (b,t) tile each.
// logits[256 x 16] = W1[256x1024] @ hsT[1024x16]; then log-softmax NLL.
// ---------------------------------------------------------------------------
__global__ void __launch_bounds__(256, 1)
k_proj(const __bf16* __restrict__ W1_bf, const float* __restrict__ b1,
       const __bf16* __restrict__ hs, const int* __restrict__ ys,
       float* __restrict__ out) {
  __shared__ __align__(16) __bf16 hsB[16 * kH];   // [n][k], 32KB
  __shared__ float logits[kV * 16];               // [v][n], 16KB
  __shared__ float red[16 * 16];                  // reduction scratch
  __shared__ float colmax[16];

  const int tid  = threadIdx.x;
  const int nt   = blockIdx.x;          // n-tile 0..1023
  const int n0   = nt * 16;
  const int wave = tid >> 5;
  const int lane = tid & 31;
  const int half = lane >> 4;
  const int m    = lane & 15;
  const int n    = lane & 15;

  stage_32k(hs + (size_t)n0 * kH, hsB, tid);      // 16 rows x 2KB contiguous
  __syncthreads();

  #pragma unroll
  for (int s = 0; s < 2; ++s) {
    const int mt = wave * 2 + s;                  // 16 M-tiles over 8 waves
    const __bf16* aptr = W1_bf + (size_t)(mt * 16 + m) * kH;
    v8f acc = {};
    #pragma unroll 4
    for (int kb = 0; kb < kH / 32; ++kb) {
      const int k0 = kb * 32;
      v16bf a = load_a_frag(aptr, k0, half);
      v16bf b = *(const v16bf*)(hsB + (size_t)n * kH + k0 + half * 16);
      acc = __builtin_amdgcn_wmma_f32_16x16x32_bf16(false, a, false, b,
                                                    (short)0, acc, false, false);
    }
    #pragma unroll
    for (int r = 0; r < 8; ++r) {
      const int row = mt * 16 + r + half * 8;
      logits[row * 16 + n] = acc[r] + b1[row];
    }
  }
  __syncthreads();

  // Parallel log-softmax NLL over the 16 columns: 16 segments x 16 rows each.
  const int col = tid & 15;
  const int seg = tid >> 4;
  float lmax = -1e30f;
  #pragma unroll
  for (int r = 0; r < 16; ++r)
    lmax = fmaxf(lmax, logits[(seg * 16 + r) * 16 + col]);
  red[seg * 16 + col] = lmax;
  __syncthreads();
  #pragma unroll
  for (int off = 8; off > 0; off >>= 1) {
    if (seg < off)
      red[seg * 16 + col] = fmaxf(red[seg * 16 + col], red[(seg + off) * 16 + col]);
    __syncthreads();
  }
  if (seg == 0) colmax[col] = red[col];
  __syncthreads();
  const float mx = colmax[col];
  float lsum = 0.0f;
  #pragma unroll
  for (int r = 0; r < 16; ++r)
    lsum += __expf(logits[(seg * 16 + r) * 16 + col] - mx);
  red[seg * 16 + col] = lsum;
  __syncthreads();
  #pragma unroll
  for (int off = 8; off > 0; off >>= 1) {
    if (seg < off)
      red[seg * 16 + col] += red[(seg + off) * 16 + col];
    __syncthreads();
  }

  if (tid < 16) {
    const int g  = n0 + tid;            // flat (b,t): g = b*T + t
    const int bb = g >> 11;
    const int tt = g & (kT - 1);
    const int y  = ys[bb * kT + tt];
    const float lse = colmax[tid] + __logf(red[tid]);
    const float nll = lse - logits[y * 16 + tid];
    atomicAdd(out, nll * (1.0f / (float)kNT));
  }
}

// ---------------------------------------------------------------------------
extern "C" void kernel_launch(void* const* d_in, const int* in_sizes, int n_in,
                              void* d_out, int out_size, void* d_ws, size_t ws_size,
                              hipStream_t stream) {
  (void)in_sizes; (void)n_in; (void)out_size;
  const int*   Xs   = (const int*)  d_in[0];
  const int*   ys   = (const int*)  d_in[1];
  /* d_in[2] = predict (0 -> loss path) */
  const float* W_ih = (const float*)d_in[3];
  const float* W_hh = (const float*)d_in[4];
  const float* b_ih = (const float*)d_in[5];
  const float* b_hh = (const float*)d_in[6];
  const float* W1   = (const float*)d_in[7];
  const float* b1   = (const float*)d_in[8];
  float* out = (float*)d_out;
  char*  ws  = (char*)d_ws;
  if (ws_size < WS_NEED) return;  // need ~41 MB of scratch

  unsigned* cnt    = (unsigned*)(ws + CNT_OFF);
  __bf16*   hbuf   = (__bf16*)(ws + HBUF_OFF);
  __bf16*   Whh_bf = (__bf16*)(ws + WHH_OFF);
  __bf16*   W1_bf  = (__bf16*)(ws + W1_OFF);
  __bf16*   hs     = (__bf16*)(ws + HS_OFF);

  k_init<<<1024, 256, 0, stream>>>(W_hh, W1, out, cnt, hbuf, Whh_bf, W1_bf);
  k_lstm<<<kNWG, 256, 0, stream>>>(Xs, W_ih, b_ih, b_hh, Whh_bf, hbuf, hs, cnt);
  k_proj<<<kNT / 16, 256, 0, stream>>>(W1_bf, b1, hs, ys, out);
}
